// MMAttn_78726750536135
// MI455X (gfx1250) — compile-verified, matrix operand order checked by reading, not run on hardware
//
#include <hip/hip_runtime.h>
#include <hip/hip_bf16.h>
#include <cstdint>

typedef __attribute__((ext_vector_type(16))) _Float16 v16h;
typedef __attribute__((ext_vector_type(8)))  _Float16 v8h;
typedef __attribute__((ext_vector_type(4)))  _Float16 v4h;
typedef __attribute__((ext_vector_type(8)))  float    v8f;

#define WMMA_F32_F16(a, b, c) \
  __builtin_amdgcn_wmma_f32_16x16x32_f16(false, (a), false, (b), (short)0, (c), false, false)

// ---------------------------------------------------------------------------
// CDNA5 async global->LDS copy (ASYNCcnt-tracked, no VGPR round trip).
// the LDS address adds the vdst VGPR and the instruction offset to the LDS base, so the VGPR operand is the
// byte offset within the workgroup LDS allocation (== addrspace(3) ptr value).
// ---------------------------------------------------------------------------
#define USE_ASYNC_LDS 1
#if USE_ASYNC_LDS
typedef __attribute__((address_space(3))) const void* lds_cptr;
__device__ __forceinline__ unsigned lds_off32(const void* p) {
  return (unsigned)(uintptr_t)(lds_cptr)p;
}
__device__ __forceinline__ void async_b128(const void* g, void* l) {
  asm volatile("global_load_async_to_lds_b128 %0, %1, off"
               :: "v"(lds_off32(l)), "v"(g) : "memory");
}
__device__ __forceinline__ void wait_async0() {
  asm volatile("s_wait_asynccnt 0" ::: "memory");
}
#endif

// Load a 16x32 f16 A/B operand fragment (symmetric layout) from LDS/memory.
// lrow = lane&15 -> matrix row (A: M, B: N); hq = lane>>4 -> K half-group.
// VGPR v holds K = (v<4?0:16) + hq*8 + (v&3)*2 + {0,1}.
__device__ __forceinline__ v16h ldfrag(const _Float16* __restrict__ p,
                                       int row, int ld, int kbase, int hq) {
  union { v16h v; unsigned u[8]; } r;
  const _Float16* rp = p + (size_t)row * ld + kbase + hq * 8;
#pragma unroll
  for (int i = 0; i < 8; ++i) {
    int k = ((i & 4) << 2) | ((i & 3) << 1);   // {0,2,4,6,16,18,20,22}
    r.u[i] = *(const unsigned*)(rp + k);
  }
  return r.v;
}

// ---------------------------------------------------------------------------
// fp32 -> f16 pack (one-time pass so GEMM staging is a pure b128 async copy).
// ---------------------------------------------------------------------------
__global__ __launch_bounds__(256) void cvt_f16(const float* __restrict__ src,
                                               _Float16* __restrict__ dst,
                                               int n4) {
  int i = blockIdx.x * 256 + threadIdx.x;
  if (i < n4) {
    float4 f = ((const float4*)src)[i];
    v4h h;
    h[0] = (_Float16)f.x; h[1] = (_Float16)f.y;
    h[2] = (_Float16)f.z; h[3] = (_Float16)f.w;
    ((v4h*)dst)[i] = h;
  }
}

// ---------------------------------------------------------------------------
// GEMM: C[M,N] = rowmap(A)[M,K] * B[N,K]^T + bias   (A,B f16; accum fp32)
// A row m -> A + (m/nseg)*a_bstride + a_off + (m%nseg)*K  (segmented batch)
// 256 threads, 128x128 block tile, K-step 32. 8 waves (4x2), wave = 32x64:
// 6 LDS fragment reads -> 8 WMMAs per K-step.
// ---------------------------------------------------------------------------
__global__ __launch_bounds__(256) void gemm_f16(
    const _Float16* __restrict__ A, const _Float16* __restrict__ B,
    const float* __restrict__ bias, float* __restrict__ C,
    int N, int K, int nseg, long long a_bstride, long long a_off) {
  __shared__ _Float16 As[128][48];
  __shared__ _Float16 Bs[128][48];
  const int tid = threadIdx.x;
  const int wave = tid >> 5, lane = tid & 31;
  const int lrow = lane & 15, hq = lane >> 4;
  const int wr = wave >> 1, wc = wave & 1;
  const int m0 = blockIdx.y * 128, n0 = blockIdx.x * 128;

  const int lr = tid >> 1;             // staged row 0..127
  const int lc = (tid & 1) << 4;       // staged col group {0,16}
  const int am = m0 + lr;
  const _Float16* arow = A + (size_t)(am / nseg) * a_bstride + a_off +
                         (size_t)(am % nseg) * K;
  const _Float16* brow = B + (size_t)(n0 + lr) * K;

  v8f acc[2][4];
#pragma unroll
  for (int i = 0; i < 2; ++i)
#pragma unroll
    for (int j = 0; j < 4; ++j)
#pragma unroll
      for (int r = 0; r < 8; ++r) acc[i][j][r] = 0.f;

  for (int k0 = 0; k0 < K; k0 += 32) {
    __syncthreads();
#if USE_ASYNC_LDS
    async_b128(arow + k0 + lc,     &As[lr][lc]);
    async_b128(arow + k0 + lc + 8, &As[lr][lc + 8]);
    async_b128(brow + k0 + lc,     &Bs[lr][lc]);
    async_b128(brow + k0 + lc + 8, &Bs[lr][lc + 8]);
    if (k0 + 32 < K) {
      __builtin_prefetch(arow + k0 + 32 + lc, 0, 0);
      __builtin_prefetch(brow + k0 + 32 + lc, 0, 0);
    }
    wait_async0();
#else
    *(v8h*)&As[lr][lc]     = *(const v8h*)(arow + k0 + lc);
    *(v8h*)&As[lr][lc + 8] = *(const v8h*)(arow + k0 + lc + 8);
    *(v8h*)&Bs[lr][lc]     = *(const v8h*)(brow + k0 + lc);
    *(v8h*)&Bs[lr][lc + 8] = *(const v8h*)(brow + k0 + lc + 8);
#endif
    __syncthreads();
    v16h af0 = ldfrag(&As[0][0], wr * 32 + lrow, 48, 0, hq);
    v16h af1 = ldfrag(&As[0][0], wr * 32 + 16 + lrow, 48, 0, hq);
#pragma unroll
    for (int j = 0; j < 4; ++j) {
      v16h bf = ldfrag(&Bs[0][0], wc * 64 + j * 16 + lrow, 48, 0, hq);
      acc[0][j] = WMMA_F32_F16(af0, bf, acc[0][j]);
      acc[1][j] = WMMA_F32_F16(af1, bf, acc[1][j]);
    }
  }
#pragma unroll
  for (int i = 0; i < 2; ++i)
#pragma unroll
    for (int j = 0; j < 4; ++j)
#pragma unroll
      for (int r = 0; r < 8; ++r) {
        int mm = m0 + wr * 32 + i * 16 + hq * 8 + r;
        int nn = n0 + wc * 64 + j * 16 + lrow;
        C[(size_t)mm * N + nn] = acc[i][j][r] + bias[nn];
      }
}

// ---------------------------------------------------------------------------
// Head split + RMSNorm + RoPE + f16 conversion. One thread per (b,h,n).
// qkv layout (from gemm): row (b-seg, n), col = t*1024 + h*64 + dd.
// Writes Q/K: (bh, 2048, 64) f16 (Q pre-scaled by 1/sqrt(64));
//        V:   (bh, 64, 2048) f16 transposed (WMMA B-operand loads in PV).
// ---------------------------------------------------------------------------
__global__ __launch_bounds__(256) void split_norm_rope(
    const float* __restrict__ C1, const float* __restrict__ C2,
    const float* __restrict__ qs1, const float* __restrict__ ks1,
    const float* __restrict__ qs2, const float* __restrict__ ks2,
    _Float16* __restrict__ Qh, _Float16* __restrict__ Kh,
    _Float16* __restrict__ Vt) {
  int idx = blockIdx.x * 256 + threadIdx.x;
  int n = idx & 2047;
  int h = (idx >> 11) & 15;
  int b = idx >> 15;
  const float* src; const float *qs, *ks;
  if (n < 1536) { src = C1 + ((size_t)b * 1536 + n) * 3072; qs = qs1; ks = ks1; }
  else          { src = C2 + ((size_t)b * 512 + (n - 1536)) * 3072; qs = qs2; ks = ks2; }
  const float pos = (float)n;
  const size_t bh = (size_t)b * 16 + h;

  for (int t = 0; t < 2; ++t) {
    const float* xr = src + t * 1024 + h * 64;
    const float* sc = t ? ks : qs;
    _Float16* out = (t ? Kh : Qh) + (bh * 2048 + n) * 64;
    float ss = 0.f;
#pragma unroll
    for (int d = 0; d < 64; ++d) ss += xr[d] * xr[d];
    float inv = rsqrtf(ss * (1.0f / 64.0f) + 1e-6f);
    float os = t ? 1.0f : 0.125f;      // fold softmax 1/sqrt(64) into Q
#pragma unroll
    for (int j = 0; j < 32; ++j) {
      float y0 = xr[2 * j] * inv * sc[2 * j];
      float y1 = xr[2 * j + 1] * inv * sc[2 * j + 1];
      float ang = pos * __expf((-2.0f * j / 64.0f) * 9.210340371976184f);
      float sn, cs;
      __sincosf(ang, &sn, &cs);
      out[2 * j]     = (_Float16)((y0 * cs - y1 * sn) * os);
      out[2 * j + 1] = (_Float16)((y0 * sn + y1 * cs) * os);
    }
  }
  const float* vr = src + 2048 + h * 64;
  _Float16* vt = Vt + bh * 64 * 2048;
#pragma unroll
  for (int d = 0; d < 64; ++d) vt[(size_t)d * 2048 + n] = (_Float16)vr[d];
}

// ---------------------------------------------------------------------------
// Flash attention. Grid (2048/128, b*h). 256 threads = 8 waves, each wave owns
// 16 Q rows. 32-wide KV tiles async-staged to LDS; online softmax with
// per-row stats in registers (rows match C-fragment layout: 8 rows per lane).
// Output written as f16 (feeds out-projection GEMM directly).
// ---------------------------------------------------------------------------
__global__ __launch_bounds__(256) void attn_flash(
    const _Float16* __restrict__ Qh, const _Float16* __restrict__ Kh,
    const _Float16* __restrict__ Vt, _Float16* __restrict__ Oh) {
  __shared__ _Float16 Kt[32][72];       // kv rows x d(64)
  __shared__ _Float16 Vl[64][48];       // d rows x kv cols(32)
  __shared__ _Float16 Pl[8][16][34];    // per-wave P staging 16x32
  const int tid = threadIdx.x;
  const int wave = tid >> 5, lane = tid & 31;
  const int lrow = lane & 15, hq = lane >> 4;
  const int bh = blockIdx.y;
  const int b = bh >> 4, h = bh & 15;
  const int q0 = blockIdx.x * 128 + wave * 16;
  const _Float16* Qb = Qh + (size_t)bh * 2048 * 64;
  const _Float16* Kb = Kh + (size_t)bh * 2048 * 64;
  const _Float16* Vb = Vt + (size_t)bh * 64 * 2048;

  v16h qf0 = ldfrag(Qb + (size_t)q0 * 64, lrow, 64, 0, hq);
  v16h qf1 = ldfrag(Qb + (size_t)q0 * 64, lrow, 64, 32, hq);

  v8f acc[4];
  float m[8], l[8];
#pragma unroll
  for (int j = 0; j < 4; ++j)
#pragma unroll
    for (int r = 0; r < 8; ++r) acc[j][r] = 0.f;
#pragma unroll
  for (int r = 0; r < 8; ++r) { m[r] = -1e30f; l[r] = 0.f; }

  const int sr = tid >> 3, scol = (tid & 7) << 3;    // K tile staging coords
  const int vr2 = tid >> 2, vc2 = (tid & 3) << 3;    // V tile staging coords

  for (int j0 = 0; j0 < 2048; j0 += 32) {
    __syncthreads();
#if USE_ASYNC_LDS
    async_b128(Kb + (size_t)(j0 + sr) * 64 + scol, &Kt[sr][scol]);
    async_b128(Vb + (size_t)vr2 * 2048 + j0 + vc2, &Vl[vr2][vc2]);
    wait_async0();
#else
    *(v8h*)&Kt[sr][scol] = *(const v8h*)(Kb + (size_t)(j0 + sr) * 64 + scol);
    *(v8h*)&Vl[vr2][vc2] = *(const v8h*)(Vb + (size_t)vr2 * 2048 + j0 + vc2);
#endif
    __syncthreads();

    v8f S0, S1;
#pragma unroll
    for (int r = 0; r < 8; ++r) { S0[r] = 0.f; S1[r] = 0.f; }
    {
      v16h kb0 = ldfrag(&Kt[0][0], lrow, 72, 0, hq);
      v16h kb1 = ldfrag(&Kt[0][0], 16 + lrow, 72, 0, hq);
      S0 = WMMA_F32_F16(qf0, kb0, S0);
      S1 = WMMA_F32_F16(qf0, kb1, S1);
      v16h kb2 = ldfrag(&Kt[0][0], lrow, 72, 32, hq);
      v16h kb3 = ldfrag(&Kt[0][0], 16 + lrow, 72, 32, hq);
      S0 = WMMA_F32_F16(qf1, kb2, S0);
      S1 = WMMA_F32_F16(qf1, kb3, S1);
    }
#pragma unroll
    for (int r = 0; r < 8; ++r) {
      float s0 = S0[r], s1 = S1[r];
      float t = fmaxf(s0, s1);                 // row reduce across 16 lanes
      t = fmaxf(t, __shfl_xor(t, 1, 32));
      t = fmaxf(t, __shfl_xor(t, 2, 32));
      t = fmaxf(t, __shfl_xor(t, 4, 32));
      t = fmaxf(t, __shfl_xor(t, 8, 32));
      float mnew = fmaxf(m[r], t);
      float scl = __expf(m[r] - mnew);
      float p0 = __expf(s0 - mnew);
      float p1 = __expf(s1 - mnew);
      float rs = p0 + p1;
      rs += __shfl_xor(rs, 1, 32);
      rs += __shfl_xor(rs, 2, 32);
      rs += __shfl_xor(rs, 4, 32);
      rs += __shfl_xor(rs, 8, 32);
      l[r] = l[r] * scl + rs;
      m[r] = mnew;
      acc[0][r] *= scl; acc[1][r] *= scl; acc[2][r] *= scl; acc[3][r] *= scl;
      Pl[wave][hq * 8 + r][lrow]      = (_Float16)p0;
      Pl[wave][hq * 8 + r][16 + lrow] = (_Float16)p1;
    }
    asm volatile("s_wait_dscnt 0" ::: "memory");   // wave-local P visibility
    v16h pf = ldfrag(&Pl[wave][0][0], lrow, 34, 0, hq);
#pragma unroll
    for (int j = 0; j < 4; ++j) {
      v16h vf = ldfrag(&Vl[0][0], j * 16 + lrow, 48, 0, hq);
      acc[j] = WMMA_F32_F16(pf, vf, acc[j]);
    }
  }
#pragma unroll
  for (int j = 0; j < 4; ++j)
#pragma unroll
    for (int r = 0; r < 8; ++r) {
      int row = q0 + hq * 8 + r;
      int col = h * 64 + j * 16 + lrow;
      Oh[((size_t)b * 2048 + row) * 1024 + col] = (_Float16)(acc[j][r] / l[r]);
    }
}

// ---------------------------------------------------------------------------
extern "C" void kernel_launch(void* const* d_in, const int* in_sizes, int n_in,
                              void* d_out, int out_size, void* d_ws, size_t ws_size,
                              hipStream_t stream) {
  const float* x1  = (const float*)d_in[0];
  const float* x2  = (const float*)d_in[1];
  const float* Wq1 = (const float*)d_in[2];
  const float* bq1 = (const float*)d_in[3];
  const float* Wq2 = (const float*)d_in[4];
  const float* bq2 = (const float*)d_in[5];
  const float* Wo1 = (const float*)d_in[6];
  const float* bo1 = (const float*)d_in[7];
  const float* Wo2 = (const float*)d_in[8];
  const float* bo2 = (const float*)d_in[9];
  const float* qs1 = (const float*)d_in[10];
  const float* ks1 = (const float*)d_in[11];
  const float* qs2 = (const float*)d_in[12];
  const float* ks2 = (const float*)d_in[13];

  char* ws = (char*)d_ws;
  float*    qkv1 = (float*)ws;                     // 3072*3072*4 = 37.75 MB
  float*    qkv2 = (float*)(ws + 37748736);        // 1024*3072*4 = 12.58 MB
  _Float16* x1h  = (_Float16*)(ws + 50331648);     // 6.29 MB
  _Float16* x2h  = (_Float16*)(ws + 56623104);     // 2.10 MB
  _Float16* Wq1h = (_Float16*)(ws + 58720256);     // 6.29 MB
  _Float16* Wq2h = (_Float16*)(ws + 65011712);     // 6.29 MB
  _Float16* Wo1h = (_Float16*)(ws + 71303168);     // 2.10 MB
  _Float16* Wo2h = (_Float16*)(ws + 73400320);     // 2.10 MB
  _Float16* Qh   = (_Float16*)(ws + 75497472);     // 8 MB
  _Float16* Kh   = (_Float16*)(ws + 83886080);     // 8 MB
  _Float16* Vt   = (_Float16*)(ws + 92274688);     // 8 MB -> end 96 MB
  _Float16* Oh   = (_Float16*)ws;                  // aliases dead qkv1 (8.4 MB)
  float*    out  = (float*)d_out;

  // 0: one-time f16 packs (removes conversions from GEMM hot loops,
  //    enables pure async b128 global->LDS staging, halves operand traffic)
  cvt_f16<<<3072, 256, 0, stream>>>(x1, x1h, 786432);
  cvt_f16<<<1024, 256, 0, stream>>>(x2, x2h, 262144);
  cvt_f16<<<3072, 256, 0, stream>>>(Wq1, Wq1h, 786432);
  cvt_f16<<<3072, 256, 0, stream>>>(Wq2, Wq2h, 786432);
  cvt_f16<<<1024, 256, 0, stream>>>(Wo1, Wo1h, 262144);
  cvt_f16<<<1024, 256, 0, stream>>>(Wo2, Wo2h, 262144);

  // 1-2: QKV projections (rows contiguous -> nseg = M, bstride = 0)
  gemm_f16<<<dim3(24, 24), 256, 0, stream>>>(x1h, Wq1h, bq1, qkv1,
                                             3072, 1024, 3072, 0, 0);
  gemm_f16<<<dim3(24, 8), 256, 0, stream>>>(x2h, Wq2h, bq2, qkv2,
                                            3072, 1024, 1024, 0, 0);
  // 3: split heads + RMSNorm + RoPE + f16 pack (+ V transpose, Q pre-scale)
  split_norm_rope<<<256, 256, 0, stream>>>(qkv1, qkv2, qs1, ks1, qs2, ks2,
                                           Qh, Kh, Vt);
  // 4: flash attention over the concatenated 2048-token sequence
  attn_flash<<<dim3(16, 32), 256, 0, stream>>>(Qh, Kh, Vt, Oh);
  // 5-6: output projections, segmented row mapping into Oh (b,2048,1024)
  gemm_f16<<<dim3(8, 24), 256, 0, stream>>>(Oh, Wo1h, bo1, out,
                                            1024, 1024, 1536,
                                            2048LL * 1024, 0);
  gemm_f16<<<dim3(8, 8), 256, 0, stream>>>(Oh, Wo2h, bo2,
                                           out + (size_t)3072 * 1024,
                                           1024, 1024, 512,
                                           2048LL * 1024, 1536LL * 1024);
}
